// CatGNN_GIN_2_17523466567802
// MI455X (gfx1250) — matrix operand order repaired and checked by reading, not compiled, not run on hardware
//
#include <hip/hip_runtime.h>
#include <hip/hip_bf16.h>

typedef __attribute__((ext_vector_type(16))) _Float16 v16h;
typedef __attribute__((ext_vector_type(8)))  _Float16 v8h;
typedef __attribute__((ext_vector_type(8)))  float    v8f;

#define NN   50000
#define EE   800000
#define GG   128
#define OUTC 10

// ---------------- elementwise helpers ----------------
__global__ void k_copy_f32(const float* __restrict__ s, float* __restrict__ d, int n) {
    int i = blockIdx.x * blockDim.x + threadIdx.x;
    if (i < n) d[i] = s[i];
}

__global__ void k_zero_f32(float* __restrict__ d, int n) {
    int i = blockIdx.x * blockDim.x + threadIdx.x;
    if (i < n) d[i] = 0.0f;
}

__global__ void k_cvt_f16(const float* __restrict__ s, _Float16* __restrict__ d, int n) {
    int i = blockIdx.x * blockDim.x + threadIdx.x;
    if (i < n) d[i] = (_Float16)s[i];
}

// ---------------- edge scatter: agg[dst] += x[src] ----------------
__global__ void k_scatter(const float* __restrict__ x, const int* __restrict__ src,
                          const int* __restrict__ dst, float* __restrict__ agg, int nE) {
    int i = blockIdx.x * blockDim.x + threadIdx.x;
    if (i >= nE * 64) return;
    int e = i >> 6, c = i & 63;
    atomicAdd(&agg[dst[e] * 64 + c], x[src[e] * 64 + c]);
}

// ---------------- global add pool: g[batch[v]] += h[v] ----------------
__global__ void k_pool(const float* __restrict__ h, const int* __restrict__ batch,
                       float* __restrict__ g, int total) {
    int i = blockIdx.x * blockDim.x + threadIdx.x;
    if (i >= total) return;
    int v = i >> 6, c = i & 63;
    atomicAdd(&g[batch[v] * 64 + c], h[i]);
}

// ---------------- column sums / sums of squares (64 columns) ----------------
__global__ void k_colstats(const float* __restrict__ h, float* __restrict__ gsum,
                           float* __restrict__ gsq, int total) {
    __shared__ float ssum[64];
    __shared__ float ssq[64];
    int t = threadIdx.x;
    if (t < 64) { ssum[t] = 0.0f; ssq[t] = 0.0f; }
    __syncthreads();
    int stride = gridDim.x * blockDim.x;   // multiple of 64 -> fixed column per thread
    float s = 0.0f, s2 = 0.0f;
    for (int i = blockIdx.x * blockDim.x + t; i < total; i += stride) {
        float v = h[i];
        s += v; s2 += v * v;
    }
    int c = t & 63;
    atomicAdd(&ssum[c], s);
    atomicAdd(&ssq[c], s2);
    __syncthreads();
    if (t < 64) {
        atomicAdd(&gsum[t], ssum[t]);
        atomicAdd(&gsq[t], ssq[t]);
    }
}

// ---------------- BN finalize: scale/shift per column ----------------
__global__ void k_bnfin(const float* __restrict__ gsum, const float* __restrict__ gsq,
                        const float* __restrict__ gamma, const float* __restrict__ beta,
                        float* __restrict__ scale, float* __restrict__ shift, float invn) {
    int c = threadIdx.x;   // 64 threads
    float mu  = gsum[c] * invn;
    float var = gsq[c] * invn - mu * mu;
    float sc  = gamma[c] * rsqrtf(var + 1e-5f);
    scale[c] = sc;
    shift[c] = beta[c] - mu * sc;
}

// ---------------- BN apply + relu + convert to f16 ----------------
__global__ void k_bnrelu_cvt(const float* __restrict__ h, const float* __restrict__ scale,
                             const float* __restrict__ shift, _Float16* __restrict__ out, int n) {
    int i = blockIdx.x * blockDim.x + threadIdx.x;
    if (i >= n) return;
    int c = i & 63;
    float v = h[i] * scale[c] + shift[c];
    v = v > 0.0f ? v : 0.0f;
    out[i] = (_Float16)v;
}

// ---------------- pack weights (f32 row-major [64 x ncols]) into WMMA B fragments ----
// Bp layout: [tn(4)][kt(2)][lane(32)][16 halves]; lane->column, VGPR halves->K.
__global__ void k_pack(const float* __restrict__ W, _Float16* __restrict__ Bp, int ncols) {
    int i = blockIdx.x * blockDim.x + threadIdx.x;
    if (i >= 4096) return;
    int h    = i & 15;
    int lane = (i >> 4) & 31;
    int kt   = (i >> 9) & 1;
    int tn   = (i >> 10) & 3;
    int n = tn * 16 + (lane & 15);
    int k = kt * 32 + ((lane >> 4) * 8) + (h < 8 ? h : h + 8);
    float v = (n < ncols) ? W[k * ncols + n] : 0.0f;
    Bp[i] = (_Float16)v;
}

// ---------------- WMMA GEMM: Out[nrows,64] = A[nrows,64] @ W[64,64] + bias (opt relu) ---
__global__ __launch_bounds__(256) void k_gemm64(
    const _Float16* __restrict__ A, const _Float16* __restrict__ Bp,
    const float* __restrict__ bias, int bias_n,
    float* __restrict__ Out, int nrows, int do_relu) {
    int wave = threadIdx.x >> 5;
    int lane = threadIdx.x & 31;
    int row0 = (blockIdx.x * 8 + wave) * 16;
    if (row0 >= nrows) return;                 // wave-uniform; active waves keep EXEC all-1

    // A fragments: 16x32 f16. lane 0-15 -> row, K in {g*8..g*8+7} U {g*8+16..g*8+23}
    int rA   = row0 + (lane & 15);
    int cgrp = (lane >> 4) * 8;
    v16h a[2];
#pragma unroll
    for (int kt = 0; kt < 2; ++kt) {
        const _Float16* p = A + rA * 64 + kt * 32 + cgrp;
        v8h lo = *(const v8h*)(p);
        v8h hi = *(const v8h*)(p + 16);
        a[kt] = __builtin_shufflevector(lo, hi, 0,1,2,3,4,5,6,7,8,9,10,11,12,13,14,15);
    }

#pragma unroll
    for (int tn = 0; tn < 4; ++tn) {
        v8f c = {};
#pragma unroll
        for (int kt = 0; kt < 2; ++kt) {
            v16h b = *(const v16h*)(Bp + (((tn * 2) + kt) * 32 + lane) * 16);
            c = __builtin_amdgcn_wmma_f32_16x16x32_f16(
                    false, a[kt], false, b, (short)0, c, false, false);
        }
        // C/D layout: VGPR r, lanes 0-15 -> M=r, N=lane; lanes 16-31 -> M=8+r, N=lane-16
        int col   = tn * 16 + (lane & 15);
        float bv  = (col < bias_n) ? bias[col] : 0.0f;
        int rbase = row0 + (lane >> 4) * 8;
#pragma unroll
        for (int r = 0; r < 8; ++r) {
            float v = c[r] + bv;
            if (do_relu) v = v > 0.0f ? v : 0.0f;
            Out[(rbase + r) * 64 + col] = v;
        }
    }
}

// ---------------- log-softmax over 10 classes per graph ----------------
__global__ void k_logsoftmax(const float* __restrict__ logits, float* __restrict__ out) {
    int r = threadIdx.x;           // 128 threads, one per graph
    if (r >= GG) return;
    const float* p = logits + r * 64;
    float m = -1e30f;
    for (int j = 0; j < OUTC; ++j) m = fmaxf(m, p[j]);
    float s = 0.0f;
    for (int j = 0; j < OUTC; ++j) s += __expf(p[j] - m);
    float ls = m + __logf(s);
    for (int j = 0; j < OUTC; ++j) out[r * OUTC + j] = p[j] - ls;
}

// =======================================================================
extern "C" void kernel_launch(void* const* d_in, const int* in_sizes, int n_in,
                              void* d_out, int out_size, void* d_ws, size_t ws_size,
                              hipStream_t stream) {
    // ---- input index resolution (insertion order vs jax sorted-key order) ----
    int idx_x, idx_edge, idx_batch, p0;
    if (in_sizes[0] == NN * 64) {            // insertion order: x, edge_index, batch, params...
        idx_x = 0; idx_edge = 1; idx_batch = 2; p0 = 3;
    } else {                                  // sorted keys: batch, edge_index, params..., x
        idx_batch = 0; idx_edge = 1; p0 = 2; idx_x = n_in - 1;
    }
    int oW1, oB1, oGa, oBe, oW2, oB2;
    if (in_sizes[p0 + 1] == 4096) {          // sorted per-conv: W1,W2,b1,b2,beta,gamma
        oW1 = 0; oW2 = 1; oB1 = 2; oB2 = 3; oBe = 4; oGa = 5;
    } else {                                  // insertion: W1,b1,gamma,beta,W2,b2
        oW1 = 0; oB1 = 1; oGa = 2; oBe = 3; oW2 = 4; oB2 = 5;
    }
    int pl = p0 + 30;                         // lin1_W, lin1_b, lin2_W, lin2_b

    const float* x_in  = (const float*)d_in[idx_x];
    const int*   edges = (const int*)d_in[idx_edge];   // jnp.int64 w/o x64 -> int32
    const int*   srcI  = edges;
    const int*   dstI  = edges + EE;
    const int*   batch = (const int*)d_in[idx_batch];

    // ---- workspace carve-up ----
    size_t off = 0;
    char* base = (char*)d_ws;
    auto carve = [&](size_t bytes) -> void* {
        void* p = base + off;
        off += (bytes + 255) & ~(size_t)255;
        return p;
    };
    float*    bufP  = (float*)carve((size_t)NN * 64 * 4);
    float*    bufQ  = (float*)carve((size_t)NN * 64 * 4);
    float*    hpre  = (float*)carve((size_t)NN * 64 * 4);
    _Float16* a16   = (_Float16*)carve((size_t)NN * 64 * 2);
    float*    stats = (float*)carve(4 * 64 * 4);   // sum | sumsq | scale | shift
    _Float16* wpack = (_Float16*)carve((size_t)12 * 4096 * 2);
    float*    gbuf  = (float*)carve((size_t)GG * 64 * 4);

    auto cdiv = [](int a, int b) { return (a + b - 1) / b; };
    const int NV = NN * 64;

    // ---- pack all weight matrices into WMMA fragment order (once per launch) ----
    for (int l = 0; l < 5; ++l) {
        k_pack<<<16, 256, 0, stream>>>((const float*)d_in[p0 + l * 6 + oW1],
                                       wpack + (size_t)(l * 2 + 0) * 4096, 64);
        k_pack<<<16, 256, 0, stream>>>((const float*)d_in[p0 + l * 6 + oW2],
                                       wpack + (size_t)(l * 2 + 1) * 4096, 64);
    }
    k_pack<<<16, 256, 0, stream>>>((const float*)d_in[pl + 0], wpack + (size_t)10 * 4096, 64);
    k_pack<<<16, 256, 0, stream>>>((const float*)d_in[pl + 2], wpack + (size_t)11 * 4096, OUTC);

    // ---- 5 GIN layers ----
    const float* xcur = x_in;
    const int gemm_blocks = cdiv(NN / 16, 8);   // 391
    for (int l = 0; l < 5; ++l) {
        float* agg = (xcur == bufP) ? bufQ : bufP;    // also the layer's output buffer
        const float* b1p = (const float*)d_in[p0 + l * 6 + oB1];
        const float* b2p = (const float*)d_in[p0 + l * 6 + oB2];
        const float* gam = (const float*)d_in[p0 + l * 6 + oGa];
        const float* bet = (const float*)d_in[p0 + l * 6 + oBe];

        // h = x + segment_sum(x[src], dst)
        k_copy_f32<<<cdiv(NV, 256), 256, 0, stream>>>(xcur, agg, NV);
        k_scatter<<<cdiv(EE * 64, 256), 256, 0, stream>>>(xcur, srcI, dstI, agg, EE);
        k_cvt_f16<<<cdiv(NV, 256), 256, 0, stream>>>(agg, a16, NV);

        // hpre = h @ W1 + b1
        k_gemm64<<<gemm_blocks, 256, 0, stream>>>(a16, wpack + (size_t)(l * 2) * 4096,
                                                  b1p, 64, hpre, NN, 0);
        // batch-norm statistics
        k_zero_f32<<<1, 256, 0, stream>>>(stats, 128);
        k_colstats<<<512, 256, 0, stream>>>(hpre, stats, stats + 64, NV);
        k_bnfin<<<1, 64, 0, stream>>>(stats, stats + 64, gam, bet,
                                      stats + 128, stats + 192, 1.0f / (float)NN);
        // relu(BN(hpre)) -> f16, then @ W2 + b2, relu
        k_bnrelu_cvt<<<cdiv(NV, 256), 256, 0, stream>>>(hpre, stats + 128, stats + 192, a16, NV);
        k_gemm64<<<gemm_blocks, 256, 0, stream>>>(a16, wpack + (size_t)(l * 2 + 1) * 4096,
                                                  b2p, 64, agg, NN, 1);
        xcur = agg;
    }

    // ---- global add pool + head ----
    float* other = ((const float*)xcur == bufP) ? bufQ : bufP;  // free big buffer for logits
    k_zero_f32<<<cdiv(GG * 64, 256), 256, 0, stream>>>(gbuf, GG * 64);
    k_pool<<<cdiv(NV, 256), 256, 0, stream>>>(xcur, batch, gbuf, NV);

    k_cvt_f16<<<cdiv(GG * 64, 256), 256, 0, stream>>>(gbuf, a16, GG * 64);
    k_gemm64<<<1, 256, 0, stream>>>(a16, wpack + (size_t)10 * 4096,
                                    (const float*)d_in[pl + 1], 64, hpre, GG, 1);
    k_cvt_f16<<<cdiv(GG * 64, 256), 256, 0, stream>>>(hpre, a16, GG * 64);
    k_gemm64<<<1, 256, 0, stream>>>(a16, wpack + (size_t)11 * 4096,
                                    (const float*)d_in[pl + 3], OUTC, other, GG, 0);
    k_logsoftmax<<<1, 128, 0, stream>>>(other, (float*)d_out);
}
